// MixtureOfExpertsModel_83176336655137
// MI455X (gfx1250) — matrix operand order repaired and checked by reading, not compiled, not run on hardware
//
#include <hip/hip_runtime.h>
#include <hip/hip_bf16.h>

typedef __attribute__((ext_vector_type(16))) __bf16 v16bf;
typedef __attribute__((ext_vector_type(4)))  __bf16 v4bf;
typedef __attribute__((ext_vector_type(8)))  float  v8f;
typedef __attribute__((ext_vector_type(4)))  float  v4f;

#define K_DIM 1024
#define KC    32
#define BM    32        // M rows per block
#define NPAD  320       // padded combined output rows (150 expert | 10 pad | 150 gate | 10 pad)
#define BS_TOT 65536

// LDS byte layout:
//   [0, 1216)        : 300 bias floats (expert | gate), padded to 304 floats
//   [1216, 6336)     : bf16 X tile, double-buffered: 2 x (32 rows x 40 halfs)
//   [6336, 47424)    : fp32 result tile, 32 rows x 321 floats
#define SB_BYTE_OFF 0
#define XS_BYTE_OFF 1216
#define XS_STRIDE_H 40
#define XS_TILE_H   (BM * XS_STRIDE_H)          // 1280 halfs = 2560 B per buffer
#define RS_BYTE_OFF (XS_BYTE_OFF + 2 * XS_TILE_H * 2)   // 6336
#define RS_STRIDE   321
#define SMEM_BYTES  (RS_BYTE_OFF + BM * RS_STRIDE * 4)  // 47424

__device__ __forceinline__ unsigned short f2bf(float f) {
    union { float f; unsigned u; } c; c.f = f;
    unsigned u = c.u;
    unsigned r = u + 0x7FFFu + ((u >> 16) & 1u);   // round-to-nearest-even
    return (unsigned short)(r >> 16);
}

// Convert 4 fp32 -> 4 packed bf16 (8 bytes) via vector convert so clang emits
// two v_cvt_pk_bf16_f32 with no repacking fix-ups.
__device__ __forceinline__ uint2 pack4_bf16(float a, float b, float c, float d) {
    v4f f = {a, b, c, d};
    v4bf h = __builtin_convertvector(f, v4bf);
    union { v4bf v; uint2 u; } u; u.v = h;
    return u.u;
}

// ---------------------------------------------------------------------------
// Kernel 1: fp32 -> bf16 weight repack into padded [320][1024] buffer in d_ws.
// rows 0..149 = expert_w, 160..309 = gate_w, others zero.
// ---------------------------------------------------------------------------
__global__ __launch_bounds__(256)
void moe_convert_w(const float* __restrict__ ew, const float* __restrict__ gw,
                   unsigned short* __restrict__ wpad) {
    int idx = blockIdx.x * 256 + threadIdx.x;     // 0 .. 320*1024-1
    int row = idx >> 10;
    int k   = idx & 1023;
    float v = 0.f;
    if (row < 150)                     v = ew[row * K_DIM + k];
    else if (row >= 160 && row < 310)  v = gw[(row - 160) * K_DIM + k];
    wpad[idx] = f2bf(v);
}

// ---------------------------------------------------------------------------
// Kernel 2: fused dual-GEMM (bf16 WMMA, fp32 accum) + softmax-combine.
// ---------------------------------------------------------------------------
__global__ __launch_bounds__(256)
void moe_kernel(const float* __restrict__ x,
                const unsigned short* __restrict__ wpad,
                const float* __restrict__ eb,
                const float* __restrict__ gb,
                float* __restrict__ out) {
    extern __shared__ char smem[];
    float*          sb = (float*)(smem + SB_BYTE_OFF);
    unsigned short* xs = (unsigned short*)(smem + XS_BYTE_OFF);
    float*          rs = (float*)(smem + RS_BYTE_OFF);

    const int tid   = threadIdx.x;
    const int lane  = tid & 31;
    const int wave  = tid >> 5;
    const int laneN = lane & 15;        // column within 16-wide tile
    const int laneH = lane >> 4;        // K half selector
    const int mtile = wave & 1;         // 2 M-tiles of 16 rows
    const int nq    = wave >> 1;        // 4 N-quarters, 5 tiles each
    const int m0    = mtile << 4;
    const size_t rowBase = (size_t)blockIdx.x * BM;

    // Stage biases (300 floats)
    for (int j = tid; j < 300; j += 256)
        sb[j] = (j < 150) ? eb[j] : gb[j - 150];

    v8f acc[5];
    const v8f vzero = {0.f, 0.f, 0.f, 0.f, 0.f, 0.f, 0.f, 0.f};
#pragma unroll
    for (int t = 0; t < 5; ++t) acc[t] = vzero;

    // X staging: 32 rows x 32 floats; thread loads one float4, converts to
    // 4 bf16, stores 8 bytes to the ping-pong LDS buffer.
    const int srow = tid >> 3;
    const int scg  = (tid & 7) << 2;
    const float* xgbase = x + (rowBase + (size_t)srow) * K_DIM + scg;
    unsigned short* xsdst = xs + srow * XS_STRIDE_H + scg;

    // A fragment source row in LDS (16-bit A layout: lane holds row m0+laneN,
    // K in {kb..kb+7} and {kb+16..kb+23}, kb = 8*laneH) -> two b128 LDS loads.
    const unsigned short* axrow = xs + (m0 + laneN) * XS_STRIDE_H;
    const int kb = laneH * 8;

    // B fragment base: lane holds bf16 of row n = 80*nq + 16*t + laneN,
    // K in {kb..kb+7} and {kb+16..kb+23}, kb = laneH*8 (matches A)
    const unsigned short* wbase =
        wpad + (size_t)(80 * nq + laneN) * K_DIM + laneH * 8;

#pragma unroll 2
    for (int k0 = 0; k0 < K_DIM; k0 += KC) {
        const int ph = ((k0 >> 5) & 1) * XS_TILE_H;   // ping-pong buffer offset

        // stage X tile: fp32 global -> bf16 LDS (converted once per block)
        float4 p = *(const float4*)(xgbase + k0);
        if (k0 + KC < K_DIM)
            __builtin_prefetch(xgbase + k0 + KC, 0, 0);   // global_prefetch_b8
        *(uint2*)(xsdst + ph) = pack4_bf16(p.x, p.y, p.z, p.w);
        __syncthreads();    // single barrier per iter; ping-pong protects reuse

        // A fragment: two 16B contiguous LDS reads (K kb..kb+7, kb+16..kb+23)
        union { uint4 q[2]; v16bf v; } au;
        au.q[0] = *(const uint4*)(axrow + ph + kb);
        au.q[1] = *(const uint4*)(axrow + ph + kb + 16);

#pragma unroll
        for (int t = 0; t < 5; ++t) {
            union { uint4 q[2]; v16bf v; } bu;
            const uint4* wq = (const uint4*)(wbase + (size_t)t * 16 * K_DIM + k0);
            bu.q[0] = wq[0];
            bu.q[1] = wq[2];
            acc[t] = __builtin_amdgcn_wmma_f32_16x16x32_bf16(
                false, au.v, false, bu.v, (short)0, acc[t], false, false);
        }
    }
    __syncthreads();

    // Spill accumulators to LDS result tile.
    // C layout: VGPR r -> M = r + 8*laneH, lane -> N = laneN
#pragma unroll
    for (int t = 0; t < 5; ++t) {
        const int col  = 16 * (nq * 5 + t) + laneN;
        const int rowb = m0 + 8 * laneH;
#pragma unroll
        for (int r = 0; r < 8; ++r)
            rs[(rowb + r) * RS_STRIDE + col] = acc[t][r];
    }
    __syncthreads();

    // Epilogue: 8 threads per row, each handles up to 2 outputs.
    const int row  = tid >> 3;
    const int part = tid & 7;
    const float* rrow = rs + row * RS_STRIDE;
#pragma unroll
    for (int oi = 0; oi < 2; ++oi) {
        const int o = part * 2 + oi;
        if (o >= 15) break;
        float miu[10], xi[10];
        float mx = -3.4e38f;
#pragma unroll
        for (int e = 0; e < 10; ++e) {
            const int c = e * 15 + o;
            miu[e] = rrow[c]       + sb[c];
            xi[e]  = rrow[160 + c] + sb[150 + c];
            mx = fmaxf(mx, xi[e]);
        }
        float ssum = 0.f, dot = 0.f;
#pragma unroll
        for (int e = 0; e < 10; ++e) {
            const float w = __expf(xi[e] - mx);
            ssum += w;
            dot  += w * miu[e];
        }
        out[(rowBase + (size_t)row) * 15 + o] = dot / ssum;
    }
}

extern "C" void kernel_launch(void* const* d_in, const int* in_sizes, int n_in,
                              void* d_out, int out_size, void* d_ws, size_t ws_size,
                              hipStream_t stream) {
    const float* x  = (const float*)d_in[0];   // (65536, 1024)
    const float* ew = (const float*)d_in[1];   // (10, 15, 1024)
    const float* eb = (const float*)d_in[2];   // (10, 15)
    const float* gw = (const float*)d_in[3];   // (10, 15, 1024)
    const float* gb = (const float*)d_in[4];   // (10, 15)
    float* out = (float*)d_out;                // (65536, 15)
    unsigned short* wpad = (unsigned short*)d_ws;  // 320*1024 bf16 = 640 KB

    // 1) repack weights to padded bf16 (L2-resident; reused by all blocks)
    moe_convert_w<<<(NPAD * K_DIM) / 256, 256, 0, stream>>>(ew, gw, wpad);

    // 2) fused dual-GEMM + softmax combine
    moe_kernel<<<BS_TOT / BM, 256, SMEM_BYTES, stream>>>(x, wpad, eb, gb, out);

    (void)in_sizes; (void)n_in; (void)out_size; (void)ws_size;
}
